// VisionAttention_21698174780127
// MI455X (gfx1250) — matrix operand order repaired and checked by reading, not compiled, not run on hardware
//
#include <hip/hip_runtime.h>

// ---------------------------------------------------------------- types ----
typedef __attribute__((ext_vector_type(16))) __bf16       bf16x16;
typedef __attribute__((ext_vector_type(8)))  float        f32x8;
typedef __attribute__((ext_vector_type(4)))  unsigned int u32x4;

union FragBF {                 // 32 bytes: one WMMA bf16 A or B operand
    bf16x16        v;
    u32x4          q[2];
    unsigned short us[16];
};

__device__ __forceinline__ f32x8 zero8() {
    f32x8 z = {0.f, 0.f, 0.f, 0.f, 0.f, 0.f, 0.f, 0.f};
    return z;
}

// fp32 -> bf16 bits, round-to-nearest-even
__device__ __forceinline__ unsigned short f2bf(float f) {
    union { float f; unsigned int u; } c; c.f = f;
    unsigned int u = c.u;
    unsigned int r = (u + 0x7FFFu + ((u >> 16) & 1u)) >> 16;
    return (unsigned short)r;
}

__device__ __forceinline__ f32x8 wmma_bf16(bf16x16 a, bf16x16 b, f32x8 c) {
    // D = A(16x32 bf16) * B(32x16 bf16) + C(16x16 f32)
    return __builtin_amdgcn_wmma_f32_16x16x32_bf16(
        /*neg_a=*/false, a, /*neg_b=*/false, b,
        /*c_mod=*/(short)0, c, /*reuse_a=*/false, /*reuse_b=*/false);
}

// ------------------------------------------------------------ constants ---
#define NB   8
#define NN   1024
#define ND   1024
#define NH   16
#define HD   64
#define ROWS (NB * NN)          // 8192

// ------------------------------------------------------- prep kernels -----
__global__ void k_cvt_bf16(const float* __restrict__ src,
                           unsigned short* __restrict__ dst, int n) {
    int i = blockIdx.x * blockDim.x + threadIdx.x;
    if (i < n) dst[i] = f2bf(src[i]);
}

// src[K][N] f32 -> dst[N][K] bf16, 32x32 LDS-tiled (256 threads)
__global__ void k_transpose_bf16(const float* __restrict__ src,
                                 unsigned short* __restrict__ dst,
                                 int K, int N) {
    __shared__ float tile[32][33];
    int kt = blockIdx.x * 32, nt = blockIdx.y * 32;
    int tx = threadIdx.x & 31, ty = threadIdx.x >> 5;   // ty in 0..7
#pragma unroll
    for (int i = 0; i < 32; i += 8)
        tile[ty + i][tx] = src[(size_t)(kt + ty + i) * N + nt + tx];
    __syncthreads();
#pragma unroll
    for (int i = 0; i < 32; i += 8)
        dst[(size_t)(nt + ty + i) * K + kt + tx] = f2bf(tile[tx][ty + i]);
}

// ------------------------------------------------ shared GEMM machinery ---
// Wave tile 32(M) x 64(N), double-buffered k-loop (k step 32).
// A: row-major bf16 [*][1024]; B^T: row-major bf16 [n][1024].
struct WaveGemm {
    const unsigned short* arow[2];   // per-M-subtile A row pointers (lane row)
    const unsigned short* brow[4];   // per-N-subtile B^T row pointers (lane col)
    int half;                        // lane>=16
    FragBF a[2][2];                  // [stage][m-subtile]
    FragBF b[2][4];                  // [stage][n-subtile]
    f32x8  acc[2][4];

    __device__ __forceinline__ void init(const unsigned short* A, int mBase,
                                         const unsigned short* Bt, int nBase,
                                         int lane) {
        half = lane >> 4;
        const int l15 = lane & 15;
#pragma unroll
        for (int u = 0; u < 2; ++u)
            arow[u] = A + (size_t)(mBase + u * 16 + l15) * 1024;
#pragma unroll
        for (int j = 0; j < 4; ++j)
            brow[j] = Bt + (size_t)(nBase + j * 16 + l15) * 1024;
#pragma unroll
        for (int u = 0; u < 2; ++u)
#pragma unroll
            for (int j = 0; j < 4; ++j) acc[u][j] = zero8();
    }

    template <int S>
    __device__ __forceinline__ void load(int k0) {
#pragma unroll
        for (int u = 0; u < 2; ++u) {
            a[S][u].q[0] = *(const u32x4*)(arow[u] + k0 + 8 * half);
            a[S][u].q[1] = *(const u32x4*)(arow[u] + k0 + 16 + 8 * half);
        }
#pragma unroll
        for (int j = 0; j < 4; ++j) {
            b[S][j].q[0] = *(const u32x4*)(brow[j] + k0 + 8 * half);
            b[S][j].q[1] = *(const u32x4*)(brow[j] + k0 + 16 + 8 * half);
        }
    }

    template <int S>
    __device__ __forceinline__ void mm() {
#pragma unroll
        for (int u = 0; u < 2; ++u)
#pragma unroll
            for (int j = 0; j < 4; ++j)
                acc[u][j] = wmma_bf16(a[S][u].v, b[S][j].v, acc[u][j]);
    }

    __device__ __forceinline__ void run() {
        load<0>(0);
        for (int k0 = 0; k0 < 1024 - 64; k0 += 64) {
            // prefetch one chunk ahead (speculative; OOB dropped by HW)
            __builtin_prefetch(arow[0] + k0 + 128, 0, 0);
            __builtin_prefetch(arow[1] + k0 + 128, 0, 0);
            __builtin_prefetch(brow[0] + k0 + 128, 0, 0);
            __builtin_prefetch(brow[2] + k0 + 128, 0, 0);
            load<1>(k0 + 32);
            mm<0>();
            load<0>(k0 + 64);
            mm<1>();
        }
        load<1>(992);
        mm<0>();            // k = 960
        mm<1>();            // k = 992
    }
};

// -------------------------------------------------- stage 1: QKV GEMM -----
// C[8192][3072] = xb @ W_qkv + b_qkv, scattered into Q (x SCALE), K, V^T.
// Block = 8 waves: 2 along M (64 rows), 4 along N (256 cols).
__global__ void __launch_bounds__(256)
k_qkv_gemm(const unsigned short* __restrict__ xb,     // [8192][1024] bf16
           const unsigned short* __restrict__ wt,     // [3072][1024] bf16 (W^T)
           const float* __restrict__ bias,            // [3072]
           unsigned short* __restrict__ Qm,           // [B*H][N][64] bf16
           unsigned short* __restrict__ Km,           // [B*H][N][64] bf16
           unsigned short* __restrict__ Vt) {         // [B*H][64][N] bf16
    const int lane = threadIdx.x & 31;
    const int w    = threadIdx.x >> 5;
    const int half = lane >> 4;
    const int l15  = lane & 15;
    const int mBase = blockIdx.x * 64 + (w & 1) * 32;
    const int nBase = blockIdx.y * 256 + (w >> 1) * 64;

    WaveGemm g;
    g.init(xb, mBase, wt, nBase, lane);
    g.run();

#pragma unroll
    for (int j = 0; j < 4; ++j) {
        const int col0 = nBase + j * 16;
        const int head = col0 / 192, rem = col0 % 192;
        const int which = rem / 64, d0 = rem % 64;
        const float bv = bias[col0 + l15];
        const int d = d0 + l15;
#pragma unroll
        for (int u = 0; u < 2; ++u) {
#pragma unroll
            for (int r = 0; r < 8; ++r) {
                const int m  = mBase + u * 16 + r + 8 * half;
                const int bb = m >> 10, n = m & 1023;
                const float val = g.acc[u][j][r] + bv;
                const size_t bh = (size_t)(bb * NH + head);
                if (which == 0)      Qm[(bh * NN + n) * HD + d] = f2bf(val * 0.125f);
                else if (which == 1) Km[(bh * NN + n) * HD + d] = f2bf(val);
                else                 Vt[(bh * HD + d) * NN + n] = f2bf(val);
            }
        }
    }
}

// ------------------------------------------- stage 2: flash attention -----
// One wave = 16 query rows of one (b,h); stream 32-key blocks with online
// softmax. S and PV both on WMMA bf16; P bounced through per-wave LDS to
// convert C-layout -> A-layout.
__global__ void __launch_bounds__(256)
k_attn(const unsigned short* __restrict__ Qm,
       const unsigned short* __restrict__ Km,
       const unsigned short* __restrict__ Vt,
       unsigned short* __restrict__ Hd) {              // [B][N][D] bf16
    __shared__ __align__(16) unsigned short pbuf[8][16 * 32];
    const int lane = threadIdx.x & 31;
    const int w    = threadIdx.x >> 5;
    const int half = lane >> 4;
    const int l15  = lane & 15;
    const int gw   = blockIdx.x * 8 + w;
    const int bh      = gw >> 6;          // B*H = 128
    const int rowTile = gw & 63;          // N/16 = 64

    const unsigned short* qp = Qm + ((size_t)bh * NN + rowTile * 16) * HD;
    const unsigned short* kp = Km + (size_t)bh * NN * HD;
    const unsigned short* vp = Vt + (size_t)bh * HD * NN;

    FragBF aq0, aq1;                       // Q rows, d = 0..31 / 32..63
    {
        const unsigned short* qrow = qp + l15 * HD;
        aq0.q[0] = *(const u32x4*)(qrow + 8 * half);
        aq0.q[1] = *(const u32x4*)(qrow + 16 + 8 * half);
        aq1.q[0] = *(const u32x4*)(qrow + 32 + 8 * half);
        aq1.q[1] = *(const u32x4*)(qrow + 48 + 8 * half);
    }

    float Mrow[8], Lrow[8];
#pragma unroll
    for (int r = 0; r < 8; ++r) { Mrow[r] = -1e30f; Lrow[r] = 0.f; }
    f32x8 o[4];
#pragma unroll
    for (int j = 0; j < 4; ++j) o[j] = zero8();

    for (int kb = 0; kb < 32; ++kb) {
        const int key0 = kb * 32;
        f32x8 s[2];
#pragma unroll
        for (int t = 0; t < 2; ++t) {      // 16 keys per sub-tile
            const unsigned short* krow = kp + (size_t)(key0 + t * 16 + l15) * HD;
            FragBF b0, b1;
            b0.q[0] = *(const u32x4*)(krow + 8 * half);
            b0.q[1] = *(const u32x4*)(krow + 16 + 8 * half);
            b1.q[0] = *(const u32x4*)(krow + 32 + 8 * half);
            b1.q[1] = *(const u32x4*)(krow + 48 + 8 * half);
            f32x8 z = zero8();
            z = wmma_bf16(aq0.v, b0.v, z);
            z = wmma_bf16(aq1.v, b1.v, z);
            s[t] = z;
        }
        // online softmax (Q pre-scaled by 1/sqrt(HD))
#pragma unroll
        for (int r = 0; r < 8; ++r) {
            float rm = fmaxf(s[0][r], s[1][r]);
            for (int msk = 1; msk < 16; msk <<= 1)
                rm = fmaxf(rm, __shfl_xor(rm, msk, 32));
            const float newM = fmaxf(Mrow[r], rm);
            const float f  = __expf(Mrow[r] - newM);
            const float p0 = __expf(s[0][r] - newM);
            const float p1 = __expf(s[1][r] - newM);
            float rs = p0 + p1;
            for (int msk = 1; msk < 16; msk <<= 1)
                rs += __shfl_xor(rs, msk, 32);
            Lrow[r] = Lrow[r] * f + rs;
            Mrow[r] = newM;
#pragma unroll
            for (int j = 0; j < 4; ++j) o[j][r] *= f;
            const int m = r + 8 * half;    // C-layout row -> LDS [m][key]
            pbuf[w][m * 32 + l15]      = f2bf(p0);
            pbuf[w][m * 32 + 16 + l15] = f2bf(p1);
        }
        // re-read P in A-layout (wave-private region; DS ops in-order)
        FragBF ap;
        ap.q[0] = *(const u32x4*)&pbuf[w][l15 * 32 + 8 * half];
        ap.q[1] = *(const u32x4*)&pbuf[w][l15 * 32 + 16 + 8 * half];
#pragma unroll
        for (int j = 0; j < 4; ++j) {      // O += P @ V  (dims j*16..)
            const unsigned short* vrow =
                vp + (size_t)(j * 16 + l15) * NN + key0 + 8 * half;
            FragBF bv;
            bv.q[0] = *(const u32x4*)(vrow);
            bv.q[1] = *(const u32x4*)(vrow + 16);
            o[j] = wmma_bf16(ap.v, bv.v, o[j]);
        }
    }

    const int bb = bh >> 4, hh = bh & 15;
#pragma unroll
    for (int j = 0; j < 4; ++j) {
#pragma unroll
        for (int r = 0; r < 8; ++r) {
            const int n_row = rowTile * 16 + r + 8 * half;
            const float val = o[j][r] / Lrow[r];
            Hd[((size_t)bb * NN + n_row) * ND + hh * HD + j * 16 + l15] = f2bf(val);
        }
    }
}

// ------------------------------------------- stage 3: output projection ---
__global__ void __launch_bounds__(256)
k_out_gemm(const unsigned short* __restrict__ Hd,     // [8192][1024] bf16
           const unsigned short* __restrict__ wot,    // [1024][1024] bf16 (W_o^T)
           const float* __restrict__ bias,            // [1024]
           float* __restrict__ out) {                 // [8192][1024] f32
    const int lane = threadIdx.x & 31;
    const int w    = threadIdx.x >> 5;
    const int half = lane >> 4;
    const int l15  = lane & 15;
    const int mBase = blockIdx.x * 64 + (w & 1) * 32;
    const int nBase = blockIdx.y * 256 + (w >> 1) * 64;

    WaveGemm g;
    g.init(Hd, mBase, wot, nBase, lane);
    g.run();

#pragma unroll
    for (int j = 0; j < 4; ++j) {
        const int col = nBase + j * 16 + l15;
        const float bv = bias[col];
#pragma unroll
        for (int u = 0; u < 2; ++u) {
#pragma unroll
            for (int r = 0; r < 8; ++r) {
                const int m = mBase + u * 16 + r + 8 * half;
                out[(size_t)m * 1024 + col] = g.acc[u][j][r] + bv;
            }
        }
    }
}

// ----------------------------------------------------------- launcher -----
extern "C" void kernel_launch(void* const* d_in, const int* in_sizes, int n_in,
                              void* d_out, int out_size, void* d_ws, size_t ws_size,
                              hipStream_t stream) {
    const float* x     = (const float*)d_in[0];   // (8,1024,1024)
    const float* W_qkv = (const float*)d_in[1];   // (1024,3072)
    const float* b_qkv = (const float*)d_in[2];   // (3072,)
    const float* W_o   = (const float*)d_in[3];   // (1024,1024)
    const float* b_o   = (const float*)d_in[4];   // (1024,)
    float* out = (float*)d_out;
    (void)in_sizes; (void)n_in; (void)out_size; (void)ws_size;

    char* ws = (char*)d_ws;
    size_t off = 0;
    auto carve = [&](size_t bytes) {
        void* p = ws + off;
        off += (bytes + 255) & ~(size_t)255;
        return p;
    };
    unsigned short* xb    = (unsigned short*)carve((size_t)ROWS * ND * 2);         // 16 MB
    unsigned short* wqkvt = (unsigned short*)carve((size_t)3 * ND * ND * 2);       //  6 MB
    unsigned short* wot   = (unsigned short*)carve((size_t)ND * ND * 2);           //  2 MB
    unsigned short* Qm    = (unsigned short*)carve((size_t)NB * NH * NN * HD * 2); // 16 MB
    unsigned short* Km    = (unsigned short*)carve((size_t)NB * NH * NN * HD * 2); // 16 MB
    unsigned short* Vt    = (unsigned short*)carve((size_t)NB * NH * HD * NN * 2); // 16 MB
    unsigned short* Hd    = (unsigned short*)carve((size_t)ROWS * ND * 2);         // 16 MB

    const int nx = ROWS * ND;
    k_cvt_bf16<<<(nx + 255) / 256, 256, 0, stream>>>(x, xb, nx);
    k_transpose_bf16<<<dim3(ND / 32, 3 * ND / 32), 256, 0, stream>>>(W_qkv, wqkvt, ND, 3 * ND);
    k_transpose_bf16<<<dim3(ND / 32, ND / 32), 256, 0, stream>>>(W_o, wot, ND, ND);

    k_qkv_gemm<<<dim3(ROWS / 64, (3 * ND) / 256), 256, 0, stream>>>(
        xb, wqkvt, b_qkv, Qm, Km, Vt);
    k_attn<<<dim3((NB * NH * (NN / 16)) / 8), 256, 0, stream>>>(Qm, Km, Vt, Hd);
    k_out_gemm<<<dim3(ROWS / 64, ND / 128 / 2), 256, 0, stream>>>(Hd, wot, b_o, out);
}
